// SpinCouplingFlow_63591285784760
// MI455X (gfx1250) — matrix operand order repaired and checked by reading, not compiled, not run on hardware
//
#include <hip/hip_runtime.h>

typedef __attribute__((ext_vector_type(16))) _Float16 v16h;
typedef __attribute__((ext_vector_type(8)))  float    v8f;
typedef __attribute__((ext_vector_type(2)))  float    v2f;
typedef __attribute__((ext_vector_type(4)))  unsigned int u32x4;

#define TWO_PI_F   6.283185307179586f
#define INV_TWO_PI 0.15915494309189535f
#define LOG2E      1.4426950408889634f
#define LN2        0.6931471805599453f

__device__ __forceinline__ float frcp(float x)  { return __builtin_amdgcn_rcpf(x); }
__device__ __forceinline__ float fexp(float x)  { return __builtin_amdgcn_exp2f(x * LOG2E); }
__device__ __forceinline__ float flog(float x)  { return __builtin_amdgcn_logf(x) * LN2; }
__device__ __forceinline__ float fsigmoid(float x) {
  return frcp(1.f + __builtin_amdgcn_exp2f(-x * LOG2E));
}
__device__ __forceinline__ float ftanh(float x) {
  float e = __builtin_amdgcn_exp2f(-2.f * LOG2E * fabsf(x));
  float r = (1.f - e) * frcp(1.f + e);
  return copysignf(r, x);
}
__device__ __forceinline__ float fsoftplus(float x) {
  // max(x,0) + log1p(exp(-|x|)), stable
  return fmaxf(x, 0.f) + flog(1.f + __builtin_amdgcn_exp2f(-fabsf(x) * LOG2E));
}
__device__ __forceinline__ float fsin(float x) { return __builtin_amdgcn_sinf(x * INV_TWO_PI); }
__device__ __forceinline__ float fcos(float x) { return __builtin_amdgcn_cosf(x * INV_TWO_PI); }
// Python-style mod into [0, 2pi)
__device__ __forceinline__ float pymod2pi(float x) {
  return x - floorf(x * INV_TWO_PI) * TWO_PI_F;
}

__device__ __forceinline__ v16h loadpair(const _Float16* p0, const _Float16* p1) {
  union { v16h h; u32x4 q[2]; } u;
  u.q[0] = *(const u32x4*)p0;   // 8 halfs, 16B aligned
  u.q[1] = *(const u32x4*)p1;   // 8 halfs, 16B aligned
  return u.h;
}

// LDS layout: 65,152 bytes (<= 64KB). h/W rows padded to 72 halfs (36 dwords)
// so that 16 lanes reading 16 different rows hit 16 distinct LDS banks.
struct __align__(16) SMem {
  float    phi[1024];        //  4096 B
  _Float16 h1[128 * 72];     // 18432 B  (offset  4096, 16B aligned)
  _Float16 h2[128 * 72];     // 18432 B  (offset 22528)
  _Float16 pr[256 * 18];     //  9216 B  (offset 40960)  params, f16
  _Float16 w2[64 * 72];      //  9216 B  (offset 50176)
  _Float16 w3[32 * 64];      //  4096 B  (offset 59392)
  union { float w1[256]; float red[256]; } u;   // 1024 B (w1 dead by reduction time)
  float b1[64]; float b2[64]; float b3[32];     // 640 B
};

__global__ __launch_bounds__(256) void spin_flow_kernel(
    const float* __restrict__ z,  const float* __restrict__ W1, const float* __restrict__ b1,
    const float* __restrict__ W2, const float* __restrict__ b2,
    const float* __restrict__ W3, const float* __restrict__ b3,
    float* __restrict__ out)
{
  __shared__ SMem s;
  const int t    = threadIdx.x;
  const int b    = blockIdx.x;
  const int lane = t & 31;
  const int wave = t >> 5;
  const int col  = lane & 15;   // N / A-row lane
  const int hi   = lane >> 4;   // lane-group within wave

  const float* zr = z + b * 1024;
  const float phi0 = zr[0];
  for (int i = t; i < 1024; i += 256) s.phi[i] = pymod2pi(zr[i] - phi0);
  float ldj = 0.f;
  __syncthreads();

  for (int layer = 0; layer < 8; ++layer) {
    // ---- stage layer weights into LDS (f16 for WMMA operands) ----
    {
      const float* g = W2 + layer * 4096;                     // [g][h] row-major
      for (int i = t; i < 4096; i += 256)
        s.w2[(i >> 6) * 72 + (i & 63)] = (_Float16)g[i];
      const float* g3 = W3 + layer * 18 * 64;                 // [p][h], pad p to 32
      for (int i = t; i < 2048; i += 256)
        s.w3[i] = (_Float16)(((i >> 6) < 18) ? g3[i] : 0.f);
      s.u.w1[t] = W1[layer * 256 + t];                        // 64x4 f32
      if (t < 64) { s.b1[t] = b1[layer * 64 + t]; s.b2[t] = b2[layer * 64 + t]; }
      if (t < 32) s.b3[t] = (t < 18) ? b3[layer * 18 + t] : 0.f;
    }
    const int par = layer & 1;
    __syncthreads();

    for (int chunk = 0; chunk < 4; ++chunk) {
      const int srow0 = chunk * 128;

      // ---- GEMM1: ctx(128x4 f32) x W1^T + b1 -> tanh -> h1 (f16)
      // V_WMMA_F32_16X16X4_F32; K=4 matches exactly. cos/sin built in-register.
      for (int i = 0; i < 4; ++i) {
        int tid = wave + 8 * i;          // 32 tiles: mtile 0..7, ntile 0..3
        int mt = tid >> 2, nt = tid & 3;
        int row = mt * 16 + col;         // chunk-local row 0..127
        int pos = 2 * (srow0 + row) + par;
        float angR = s.phi[(pos + 1) & 1023];
        float angL = s.phi[(pos + 1023) & 1023];
        float cR = fcos(angR), sR = fsin(angR);
        float cL = fcos(angL), sL = fsin(angL);
        // A 16x4 f32: lanes0-15 -> K0,K2 ; lanes16-31 -> K1,K3 ; ctx=[cosR,sinR,cosL,sinL]
        v2f a; a.x = hi ? sR : cR; a.y = hi ? sL : cL;
        // B 4x16 f32: v0 = rows K0(l0-15)/K1(l16-31); v1 = K2/K3 ; B[k][n]=W1[n][k]
        int wrow = nt * 16 + col;
        v2f bb; bb.x = s.u.w1[wrow * 4 + hi]; bb.y = s.u.w1[wrow * 4 + 2 + hi];
        float cb = s.b1[wrow];
        v8f acc = {cb, cb, cb, cb, cb, cb, cb, cb};
        acc = __builtin_amdgcn_wmma_f32_16x16x4_f32(
            false, a, false, bb, (short)0, acc, false, false);
        #pragma unroll
        for (int r = 0; r < 8; ++r) {
          int ro = mt * 16 + r + 8 * hi;                   // C/D: M=r (+8 for hi lanes)
          s.h1[ro * 72 + nt * 16 + col] = (_Float16)ftanh(acc[r]);
        }
      }
      __syncthreads();

      // ---- GEMM2: h1(128x64) x W2^T + b2 -> tanh -> h2 ; 2x V_WMMA_F32_16X16X32_F16
      for (int i = 0; i < 4; ++i) {
        int tid = wave + 8 * i;
        int mt = tid >> 2, nt = tid & 3;
        int arow = mt * 16 + col;
        int wrow = nt * 16 + col;
        float cb = s.b2[wrow];
        v8f acc = {cb, cb, cb, cb, cb, cb, cb, cb};
        #pragma unroll
        for (int kt = 0; kt < 64; kt += 32) {
          // A 16x32 f16: lane row=arow; K = 16*hi .. +15 contiguous
          const _Float16* pa = &s.h1[arow * 72 + kt + 16 * hi];
          v16h av = loadpair(pa, pa + 8);
          // B 32x16 f16: col=lane&15; K = 16*hi .. +15 contiguous ; B[k][n]=W2[n][k]
          const _Float16* pb = &s.w2[wrow * 72 + kt + 16 * hi];
          v16h bv = loadpair(pb, pb + 8);
          acc = __builtin_amdgcn_wmma_f32_16x16x32_f16(
              false, av, false, bv, (short)0, acc, false, false);
        }
        #pragma unroll
        for (int r = 0; r < 8; ++r) {
          int ro = mt * 16 + r + 8 * hi;
          s.h2[ro * 72 + nt * 16 + col] = (_Float16)ftanh(acc[r]);
        }
      }
      __syncthreads();

      // ---- GEMM3: h2 x W3^T + b3 -> params (18 cols, padded to 32)
      for (int i = 0; i < 2; ++i) {
        int tid = wave + 8 * i;          // 16 tiles: mtile 0..7, ntile 0..1
        int mt = tid >> 1, nt = tid & 1;
        int arow = mt * 16 + col;
        int wrow = nt * 16 + col;
        float cb = s.b3[wrow];
        v8f acc = {cb, cb, cb, cb, cb, cb, cb, cb};
        #pragma unroll
        for (int kt = 0; kt < 64; kt += 32) {
          const _Float16* pa = &s.h2[arow * 72 + kt + 16 * hi];
          v16h av = loadpair(pa, pa + 8);
          const _Float16* pb = &s.w3[wrow * 64 + kt + 16 * hi];
          v16h bv = loadpair(pb, pb + 8);
          acc = __builtin_amdgcn_wmma_f32_16x16x32_f16(
              false, av, false, bv, (short)0, acc, false, false);
        }
        int c_ = nt * 16 + col;
        if (c_ < 18) {
          #pragma unroll
          for (int r = 0; r < 8; ++r) {
            int ro = (chunk & 1) * 128 + mt * 16 + r + 8 * hi;
            s.pr[ro * 18 + c_] = (_Float16)acc[r];
          }
        }
      }
      __syncthreads();

      // ---- elementwise spline update: once per 2 chunks (256 slots, 1/thread)
      if (chunk & 1) {
        int sidx = (chunk - 1) * 128 + t;     // slot 0..511
        int pos  = 2 * sidx + par;
        bool active = !(par == 0 && sidx == 0);   // even layers exclude position 0
        float yv = 0.f;
        if (active) {
          const _Float16* pp = &s.pr[t * 18];
          float av6[6], sh6[6], wv6[6];
          float sum = 0.f;
          #pragma unroll
          for (int j = 0; j < 6; ++j) {
            av6[j] = fsoftplus((float)pp[j]) + 0.001f;
            sh6[j] = (float)pp[6 + j];
            float e = fexp((float)pp[12 + j]);
            wv6[j] = e; sum += e;
          }
          float winv = frcp(sum);
          float x  = s.phi[pos] * INV_TWO_PI;
          float ox = 1.f - x;
          float xp = x * x, oxp = ox * ox;
          float iden = frcp(xp + oxp);
          float u  = xp * iden;
          float du = 2.f * x * ox * iden * iden;
          float y = 0.f, dy = 0.f;
          #pragma unroll
          for (int j = 0; j < 6; ++j) {
            float wj = 0.01f + 0.94f * wv6[j] * winv;
            float s1 = fsigmoid(av6[j] + sh6[j]);
            float s0 = fsigmoid(sh6[j]);
            float sv = fsigmoid(av6[j] * u + sh6[j]);
            float idn = frcp(s1 - s0);
            y  += wj * (sv - s0) * idn;
            dy += wj * av6[j] * sv * (1.f - sv) * idn;
          }
          dy *= du;
          ldj += flog(dy);
          yv = y * TWO_PI_F;
        }
        __syncthreads();
        if (active) s.phi[pos] = yv;
        __syncthreads();
      }
    } // chunks
    __syncthreads();
  } // layers

  // ---- outputs: phi (B*L floats) then ldj (B floats)
  for (int i = t; i < 1024; i += 256)
    out[b * 1024 + i] = pymod2pi(s.phi[i] + phi0);
  __syncthreads();
  s.u.red[t] = ldj;
  __syncthreads();
  for (int off = 128; off > 0; off >>= 1) {
    if (t < off) s.u.red[t] += s.u.red[t + off];
    __syncthreads();
  }
  if (t == 0) out[1024 * 1024 + b] = s.u.red[0];
}

extern "C" void kernel_launch(void* const* d_in, const int* in_sizes, int n_in,
                              void* d_out, int out_size, void* d_ws, size_t ws_size,
                              hipStream_t stream) {
  const float* z  = (const float*)d_in[0];
  const float* W1 = (const float*)d_in[1];
  const float* b1 = (const float*)d_in[2];
  const float* W2 = (const float*)d_in[3];
  const float* b2 = (const float*)d_in[4];
  const float* W3 = (const float*)d_in[5];
  const float* b3 = (const float*)d_in[6];
  spin_flow_kernel<<<dim3(1024), dim3(256), 0, stream>>>(
      z, W1, b1, W2, b2, W3, b3, (float*)d_out);
}
